// FaceWarpingLayer_16286515986674
// MI455X (gfx1250) — matrix operand order repaired and checked by reading, not compiled, not run on hardware
//
#include <hip/hip_runtime.h>
#include <hip/hip_bf16.h>

typedef __attribute__((ext_vector_type(16))) _Float16 v16h;
typedef __attribute__((ext_vector_type(8)))  float    v8f;

#define HW    (512 * 512)
#define NTRI  98
#define NB    8

// ---------------------------------------------------------------------------
// Kernel 1: per-(batch, triangle) affine solve.
//   P = [[ax,ay,1],[bx,by,1],[cx,cy,1]] (template verts), src = warped verts.
//   X = P^{-1} src (3x2), theta[d][j] = X[j][d].  Stored padded to 8 floats
//   so the sampler can do two aligned float4 loads.
// ---------------------------------------------------------------------------
__global__ void theta_solve_kernel(const float* __restrict__ pts,
                                   const float* __restrict__ tmpl,
                                   const int*   __restrict__ tri,
                                   float* __restrict__ thetaTab) {
    int id = blockIdx.x * blockDim.x + threadIdx.x;
    if (id >= NB * NTRI) return;
    int b = id / NTRI, t = id % NTRI;

    int i0 = tri[t * 3 + 0], i1 = tri[t * 3 + 1], i2 = tri[t * 3 + 2];
    float ax = tmpl[i0 * 2], ay = tmpl[i0 * 2 + 1];
    float bx = tmpl[i1 * 2], by = tmpl[i1 * 2 + 1];
    float cx = tmpl[i2 * 2], cy = tmpl[i2 * 2 + 1];

    float det = ax * (by - cy) - ay * (bx - cx) + (bx * cy - by * cx);
    float rd  = 1.0f / det;
    // inv(P) via adjugate
    float i00 = (by - cy) * rd, i01 = (cy - ay) * rd, i02 = (ay - by) * rd;
    float i10 = (cx - bx) * rd, i11 = (ax - cx) * rd, i12 = (bx - ax) * rd;
    float i20 = (bx * cy - by * cx) * rd;
    float i21 = (ay * cx - ax * cy) * rd;
    float i22 = (ax * by - ay * bx) * rd;

    float s0x = pts[(b * 64 + i0) * 2], s0y = pts[(b * 64 + i0) * 2 + 1];
    float s1x = pts[(b * 64 + i1) * 2], s1y = pts[(b * 64 + i1) * 2 + 1];
    float s2x = pts[(b * 64 + i2) * 2], s2y = pts[(b * 64 + i2) * 2 + 1];

    float* o = thetaTab + (size_t)id * 8;
    // theta row 0 (x' coeffs): X[j][0] = inv[j][:] . (s0x,s1x,s2x)
    o[0] = i00 * s0x + i01 * s1x + i02 * s2x;
    o[1] = i10 * s0x + i11 * s1x + i12 * s2x;
    o[2] = i20 * s0x + i21 * s1x + i22 * s2x;
    // theta row 1 (y' coeffs)
    o[3] = i00 * s0y + i01 * s1y + i02 * s2y;
    o[4] = i10 * s0y + i11 * s1y + i12 * s2y;
    o[5] = i20 * s0y + i21 * s1y + i22 * s2y;
    o[6] = 0.0f; o[7] = 0.0f;
}

// ---------------------------------------------------------------------------
// Kernel 2: fused owner-selection (WMMA over tri_masks) + affine warp +
// bilinear grid_sample.  One wave32 per 16-pixel tile; 8 waves / block.
// ---------------------------------------------------------------------------
__global__ __launch_bounds__(256)
void face_warp_kernel(const float* __restrict__ img,
                      const float* __restrict__ masks,
                      const float* __restrict__ thetaTab,
                      float* __restrict__ out) {
    const int lane = threadIdx.x & 31;
    const int wv   = threadIdx.x >> 5;
    const int m16  = lane & 15;       // A-matrix row / B-matrix column index
    const int hi   = lane >> 4;       // upper half-wave flag
    const int p0   = blockIdx.x * 128 + wv * 16;   // tile's first pixel
    const int p    = p0 + m16;                     // this lane's A-row pixel

    // owner+1 = tri_masks^T @ (t+1): exact in f16 (masks are 0/1, t+1 <= 98)
    v8f acc = {0.f, 0.f, 0.f, 0.f, 0.f, 0.f, 0.f, 0.f};
    #pragma unroll
    for (int ks = 0; ks < 4; ++ks) {
        if (ks < 3)  // prefetch next 32-triangle slice of this pixel column
            __builtin_prefetch(&masks[(size_t)((ks + 1) * 32) * HW + p], 0, 1);
        v16h a, bb;
        #pragma unroll
        for (int s = 0; s < 16; ++s) {
            // A (16x32 f16) slot->K map per ISA 7.12.2
            int kA = ((((s >> 1) & 3) << 1) + (s & 1)) + ((s >= 8) ? 16 : 0) + (hi ? 8 : 0);
            int tA = ks * 32 + kA;
            float mv = (tA < NTRI) ? masks[(size_t)tA * HW + p] : 0.0f;
            a[s] = (_Float16)mv;
            // B (32x16 f16): same slot->K map as A
            int tB = ks * 32 + (s & 7) + ((s >= 8) ? 16 : 0) + (hi ? 8 : 0);
            float bv = ((m16 == 0) && (tB < NTRI)) ? (float)(tB + 1) : 0.0f;
            bb[s] = (_Float16)bv;
        }
        acc = __builtin_amdgcn_wmma_f32_16x16x32_f16(
            false, a, false, bb, (short)0, acc, false, false);
    }

    // Column N=0 of D lives in lanes 0 (rows 0-7) and 16 (rows 8-15).
    __shared__ float ldsOwn[8][16];
    if (m16 == 0) {
        #pragma unroll
        for (int r = 0; r < 8; ++r) ldsOwn[wv][hi * 8 + r] = acc[r];
    }
    __syncthreads();

    // Sampling: 16 pixels x 8 batches = 128 tasks per wave, 4 per lane.
    #pragma unroll
    for (int k = 0; k < 4; ++k) {
        const int b  = 2 * k + hi;
        const int pp = p0 + m16;
        const int h  = pp >> 9, w = pp & 511;
        const int ow = (int)(ldsOwn[wv][m16] + 0.5f) - 1;

        float o0 = 0.f, o1 = 0.f, o2 = 0.f;
        if (ow >= 0) {
            const float4* th = (const float4*)(thetaTab + ((size_t)(b * NTRI + ow) << 3));
            float4 u = th[0], v = th[1];
            float gx = (w + 0.5f) * (2.0f / 512.0f) - 1.0f;
            float gy = (h + 0.5f) * (2.0f / 512.0f) - 1.0f;
            float xg = u.x * gx + u.y * gy + u.z;
            float yg = u.w * gx + v.x * gy + v.y;

            float x = (xg + 1.0f) * 256.0f - 0.5f;
            float y = (yg + 1.0f) * 256.0f - 0.5f;
            float x0f = floorf(x), y0f = floorf(y);
            float wx = x - x0f, wy = y - y0f;
            float x1f = x0f + 1.0f, y1f = y0f + 1.0f;
            float vx0 = (x0f >= 0.f && x0f < 512.f) ? 1.f : 0.f;
            float vx1 = (x1f >= 0.f && x1f < 512.f) ? 1.f : 0.f;
            float vy0 = (y0f >= 0.f && y0f < 512.f) ? 1.f : 0.f;
            float vy1 = (y1f >= 0.f && y1f < 512.f) ? 1.f : 0.f;
            int ix0 = (int)fminf(fmaxf(x0f, 0.f), 511.f);
            int ix1 = (int)fminf(fmaxf(x1f, 0.f), 511.f);
            int iy0 = (int)fminf(fmaxf(y0f, 0.f), 511.f);
            int iy1 = (int)fminf(fmaxf(y1f, 0.f), 511.f);

            float w00 = (1.f - wx) * (1.f - wy), w01 = wx * (1.f - wy);
            float w10 = (1.f - wx) * wy,         w11 = wx * wy;

            const float* ib = img + (size_t)b * 3 * HW;
            #pragma unroll
            for (int c = 0; c < 3; ++c) {
                const float* pc = ib + (size_t)c * HW;
                float v00 = pc[iy0 * 512 + ix0] * vx0 * vy0;
                float v01 = pc[iy0 * 512 + ix1] * vx1 * vy0;
                float v10 = pc[iy1 * 512 + ix0] * vx0 * vy1;
                float v11 = pc[iy1 * 512 + ix1] * vx1 * vy1;
                float r = v00 * w00 + v01 * w01 + v10 * w10 + v11 * w11;
                if (c == 0) o0 = r; else if (c == 1) o1 = r; else o2 = r;
            }
        }
        out[(size_t)(b * 3 + 0) * HW + pp] = o0;
        out[(size_t)(b * 3 + 1) * HW + pp] = o1;
        out[(size_t)(b * 3 + 2) * HW + pp] = o2;
    }
}

extern "C" void kernel_launch(void* const* d_in, const int* in_sizes, int n_in,
                              void* d_out, int out_size, void* d_ws, size_t ws_size,
                              hipStream_t stream) {
    const float* img   = (const float*)d_in[0];   // (8,3,512,512)
    const float* pts   = (const float*)d_in[1];   // (8,64,2)
    const float* tmpl  = (const float*)d_in[2];   // (64,2)
    const int*   tri   = (const int*)  d_in[3];   // (98,3)
    const float* masks = (const float*)d_in[4];   // (98,512,512)
    // d_in[5] = face_mask — implied by owner map, unused.

    float* thetaTab = (float*)d_ws;               // 8*98*8 floats = 25 KB
    float* outf     = (float*)d_out;

    theta_solve_kernel<<<4, 256, 0, stream>>>(pts, tmpl, tri, thetaTab);
    face_warp_kernel<<<HW / 128, 256, 0, stream>>>(img, masks, thetaTab, outf);
}
